// GAT_90211493085597
// MI455X (gfx1250) — matrix operand order repaired and checked by reading, not compiled, not run on hardware
//
#include <hip/hip_runtime.h>
#include <hip/hip_bf16.h>
#include <math.h>

// Problem constants (match reference)
#define NNODES   100000
#define NEDGES   1600000
#define ETOT     (NEDGES + NNODES)   // edges + self loops
#define HID      64                  // F == H == 64
#define NCLS     10
#define NGRAPH   128
#define NEG_SLOPE 0.2f

typedef __attribute__((ext_vector_type(16))) _Float16 v16h;
typedef __attribute__((ext_vector_type(8)))  float    v8f;

// ---------- order-preserving float <-> uint encoding for atomic max ----------
__device__ __forceinline__ unsigned floatFlip(float f) {
  unsigned u = __float_as_uint(f);
  return (u & 0x80000000u) ? ~u : (u | 0x80000000u);
}
__device__ __forceinline__ float floatUnflip(unsigned u) {
  return __uint_as_float((u & 0x80000000u) ? (u & 0x7fffffffu) : ~u);
}
#define NEG_INF_FLIPPED 0x007FFFFFu   // floatFlip(-inf)

// ---------------------------------------------------------------------------
// Y[nrows x HOUT] = act( X[nrows x 64] @ W[64 x HOUT] + bias )
//
// 256 threads = 8 wave32. W (64 x HOUT, zero-padded to 64 x NC) is staged once
// per block into LDS as f16, transposed to [col][k] so each lane's B fragment
// is two 8-half runs 16 k apart (2x ds_read_b128). Each wave owns a 16-row
// strip and computes all NT column tiles (A-fragment reuse): K=64 fixed ->
// 2 x NT v_wmma_f32_16x16x32_f16 per wave. A fragments load as float4
// (global_load_b128) + v_cvt_pk_f16_f32; no divergence anywhere near WMMA.
// ACT: 0=none, 1=relu, 2=elu. BIAS: 0/1.
// ---------------------------------------------------------------------------
template<int HOUT, int ACT, int BIAS>
__global__ void __launch_bounds__(256)
gemm64_wmma(const float* __restrict__ X, const float* __restrict__ W,
            const float* __restrict__ bias, float* __restrict__ Y, int nrows)
{
  constexpr int NT  = (HOUT + 15) / 16;   // column tiles (4 or 1)
  constexpr int NC  = NT * 16;            // padded column count (64 or 16)
  constexpr int LDW = 72;                 // halves per column (64 + 8 pad), 144B (16B-mult)
  __shared__ _Float16 ldsW[NC * LDW];

  const int tid = threadIdx.x;

  // ---- cooperative staging of W into LDS (f16, transposed, zero-padded) ----
  #pragma unroll
  for (int idx = tid; idx < NC * 64; idx += 256) {
    const int col = idx & (NC - 1);                 // NC is a power of two
    const int k   = idx >> (NT == 1 ? 4 : 6);
    const float w = (col < HOUT) ? W[k * HOUT + col] : 0.0f;
    ldsW[col * LDW + k] = (_Float16)w;
  }
  __syncthreads();

  const int lane = tid & 31;
  const int wv   = tid >> 5;                        // wave id in block: 0..7
  const int row0 = (blockIdx.x * 8 + wv) * 16;
  if (row0 >= nrows) return;                        // wave-uniform guard

  const int half = lane >> 4;                       // 0: lanes 0-15, 1: lanes 16-31
  const int l15  = lane & 15;
  const int arow = row0 + l15;                      // A fragment: M index of this lane

  v8f acc[NT];
  #pragma unroll
  for (int t = 0; t < NT; ++t) acc[t] = (v8f){};

  const float4* __restrict__ Xv = (const float4*)(X + arow * 64);

  #pragma unroll
  for (int kk = 0; kk < 64; kk += 32) {
    // A 16x32 f16 fragment (ISA 7.12.2): halves 0-7 -> K = kk+half*8+j,
    // halves 8-15 -> K = kk+16+half*8+(j-8); both runs are contiguous floats.
    const int q0 = (kk + half * 8) >> 2;            // float4 index of first run
    const int q1 = (kk + 16 + half * 8) >> 2;       // float4 index of second run
    const float4 p0 = Xv[q0], p1 = Xv[q0 + 1];
    const float4 p2 = Xv[q1], p3 = Xv[q1 + 1];
    v16h a;
    a[0]=(_Float16)p0.x;  a[1]=(_Float16)p0.y;  a[2]=(_Float16)p0.z;  a[3]=(_Float16)p0.w;
    a[4]=(_Float16)p1.x;  a[5]=(_Float16)p1.y;  a[6]=(_Float16)p1.z;  a[7]=(_Float16)p1.w;
    a[8]=(_Float16)p2.x;  a[9]=(_Float16)p2.y;  a[10]=(_Float16)p2.z; a[11]=(_Float16)p2.w;
    a[12]=(_Float16)p3.x; a[13]=(_Float16)p3.y; a[14]=(_Float16)p3.z; a[15]=(_Float16)p3.w;

    #pragma unroll
    for (int t = 0; t < NT; ++t) {
      // B 32x16 f16 fragment: lane = column, halves 0-7 -> K kk+half*8+j, 8-15 -> K kk+16+half*8+(j-8)
      const _Float16* __restrict__ wp = &ldsW[(t * 16 + l15) * LDW + kk + half * 8];
      v16h b;
      #pragma unroll
      for (int j = 0; j < 16; ++j) b[j] = wp[j + (j >= 8 ? 8 : 0)];
      acc[t] = __builtin_amdgcn_wmma_f32_16x16x32_f16(false, a, false, b,
                                                      (short)0, acc[t], false, false);
    }
  }

  // ---- epilogue: D f32 layout: VGPR r holds M = r + 8*half, N = lane&15 ----
  #pragma unroll
  for (int t = 0; t < NT; ++t) {
    const int bcol = t * 16 + l15;
    if (bcol < HOUT) {                              // only live for HOUT=10 tail
      const float bv = BIAS ? bias[bcol] : 0.0f;
      #pragma unroll
      for (int r = 0; r < 8; ++r) {
        float v = acc[t][r] + bv;
        if (ACT == 1)      v = v > 0.0f ? v : 0.0f;
        else if (ACT == 2) v = v > 0.0f ? v : expm1f(v);
        Y[(row0 + half * 8 + r) * HOUT + bcol] = v;
      }
    }
  }
}

// ---------------------------------------------------------------------------
// Per-node attention halves: al_s[n] = h[n,:]·a_s ; al_d[n] = h[n,:]·a_d
// ---------------------------------------------------------------------------
__global__ void attn_logits(const float* __restrict__ h,
                            const float* __restrict__ a_s,
                            const float* __restrict__ a_d,
                            float* __restrict__ al_s, float* __restrict__ al_d)
{
  int n = blockIdx.x * blockDim.x + threadIdx.x;
  if (n >= NNODES) return;
  const float4* __restrict__ hv = (const float4*)(h + n * HID);
  const float4* __restrict__ sv = (const float4*)a_s;
  const float4* __restrict__ dv = (const float4*)a_d;
  float ss = 0.0f, sd = 0.0f;
  #pragma unroll
  for (int q = 0; q < HID / 4; ++q) {
    float4 x = hv[q], s = sv[q], d = dv[q];
    ss += x.x * s.x + x.y * s.y + x.z * s.z + x.w * s.w;
    sd += x.x * d.x + x.y * d.y + x.z * d.z + x.w * d.w;
  }
  al_s[n] = ss;
  al_d[n] = sd;
}

// ---------------------------------------------------------------------------
// Init: zero aggregation buffer + denom, set segment-max to -inf; optionally
// zero the final output (first layer only).
// ---------------------------------------------------------------------------
__global__ void init_buffers(float* __restrict__ agg, float* __restrict__ denom,
                             unsigned* __restrict__ m_enc,
                             float* __restrict__ dout, int zero_dout)
{
  int i = blockIdx.x * blockDim.x + threadIdx.x;
  if (i < NNODES * HID) agg[i] = 0.0f;
  if (i < NNODES) { denom[i] = 0.0f; m_enc[i] = NEG_INF_FLIPPED; }
  if (zero_dout && i < NGRAPH * NCLS) dout[i] = 0.0f;
}

__device__ __forceinline__ void edge_pair(const int* __restrict__ ei, int e,
                                          int& s, int& d)
{
  if (e < NEDGES) { s = ei[e]; d = ei[NEDGES + e]; }
  else            { s = e - NEDGES; d = s; }           // implicit self loop
}

// Pass 1: e = leaky_relu(al_s[src] + al_d[dst]); segment max over dst (atomic).
__global__ void edge_logit_max(const int* __restrict__ ei,
                               const float* __restrict__ al_s,
                               const float* __restrict__ al_d,
                               float* __restrict__ eval,
                               unsigned* __restrict__ m_enc)
{
  int e = blockIdx.x * blockDim.x + threadIdx.x;
  if (e >= ETOT) return;
  int s, d; edge_pair(ei, e, s, d);
  float v = al_s[s] + al_d[d];
  v = v > 0.0f ? v : NEG_SLOPE * v;
  eval[e] = v;
  atomicMax(&m_enc[d], floatFlip(v));
}

// Pass 2: ex = exp(e - m[dst]); segment sum into denom.
__global__ void edge_exp_sum(const int* __restrict__ ei,
                             float* __restrict__ eval,
                             const unsigned* __restrict__ m_enc,
                             float* __restrict__ denom)
{
  int e = blockIdx.x * blockDim.x + threadIdx.x;
  if (e >= ETOT) return;
  int d = (e < NEDGES) ? ei[NEDGES + e] : (e - NEDGES);
  float ex = expf(eval[e] - floatUnflip(m_enc[d]));
  eval[e] = ex;
  atomicAdd(&denom[d], ex);
}

// Pass 3: one wave per edge; agg[dst,:] += (ex/denom[dst]) * h[src,:]
__global__ void edge_scatter(const int* __restrict__ ei,
                             const float* __restrict__ eval,
                             const float* __restrict__ denom,
                             const float* __restrict__ h,
                             float* __restrict__ agg)
{
  int gtid = blockIdx.x * blockDim.x + threadIdx.x;
  int e    = gtid >> 5;                 // wave id == edge id (uniform per wave)
  int lane = threadIdx.x & 31;
  if (e >= ETOT) return;
  int s, d; edge_pair(ei, e, s, d);
  float alpha = eval[e] / denom[d];
  #pragma unroll
  for (int f = lane; f < HID; f += 32)
    atomicAdd(&agg[d * HID + f], alpha * h[s * HID + f]);
}

// Post-aggregation epilogue: y = elu(y + bias)
__global__ void bias_elu(float* __restrict__ y, const float* __restrict__ b)
{
  int i = blockIdx.x * blockDim.x + threadIdx.x;
  if (i >= NNODES * HID) return;
  float v = y[i] + b[i & (HID - 1)];
  y[i] = v > 0.0f ? v : expm1f(v);
}

// global_add_pool: out[batch[n], c] += logits[n, c]
__global__ void pool_kernel(const float* __restrict__ logits,
                            const int* __restrict__ batch,
                            float* __restrict__ out)
{
  int n = blockIdx.x * blockDim.x + threadIdx.x;
  if (n >= NNODES) return;
  int g = batch[n];
  #pragma unroll
  for (int c = 0; c < NCLS; ++c)
    atomicAdd(&out[g * NCLS + c], logits[n * NCLS + c]);
}

// ---------------------------------------------------------------------------
extern "C" void kernel_launch(void* const* d_in, const int* in_sizes, int n_in,
                              void* d_out, int out_size, void* d_ws, size_t ws_size,
                              hipStream_t stream)
{
  (void)in_sizes; (void)n_in; (void)out_size; (void)ws_size;

  // Inputs in setup_inputs() order
  const float* x     = (const float*)d_in[0];
  const int*   ei    = (const int*)  d_in[1];   // [2, E] flat: src row then dst row
  const int*   batch = (const int*)  d_in[2];
  const float* W1    = (const float*)d_in[3];
  const float* as1   = (const float*)d_in[4];
  const float* ad1   = (const float*)d_in[5];
  const float* b1    = (const float*)d_in[6];
  const float* W2    = (const float*)d_in[7];
  const float* as2   = (const float*)d_in[8];
  const float* ad2   = (const float*)d_in[9];
  const float* b2    = (const float*)d_in[10];
  const float* mw1   = (const float*)d_in[11];
  const float* mb1   = (const float*)d_in[12];
  const float* mw2   = (const float*)d_in[13];
  const float* mb2   = (const float*)d_in[14];
  float* out = (float*)d_out;                   // [G, C] float32

  // Workspace carve-up (~60 MB of floats)
  float*    bufA  = (float*)d_ws;               // N*HID  (h = X@W)
  float*    bufB  = bufA + NNODES * HID;        // N*HID  (aggregation / next input)
  float*    eval  = bufB + NNODES * HID;        // ETOT   (edge vals; reused as logits N*C)
  float*    al_s  = eval + ETOT;                // N
  float*    al_d  = al_s + NNODES;              // N
  unsigned* m_enc = (unsigned*)(al_d + NNODES); // N
  float*    denom = (float*)(m_enc + NNODES);   // N

  const int T = 256;
  const int gGemm  = (NNODES + 127) / 128;                // 8 wave-strips of 16 rows
  const int gInit  = (NNODES * HID + T - 1) / T;
  const int gNode  = (NNODES + T - 1) / T;
  const int gEdge  = (ETOT + T - 1) / T;
  const int gScat  = (ETOT * 32 + T - 1) / T;             // one wave32 per edge

  // ---- GAT layer 1: F(64) -> H(64) ----
  init_buffers<<<gInit, T, 0, stream>>>(bufB, denom, m_enc, out, 1);
  gemm64_wmma<HID, 0, 0><<<gGemm, T, 0, stream>>>(x, W1, nullptr, bufA, NNODES);
  attn_logits  <<<gNode, T, 0, stream>>>(bufA, as1, ad1, al_s, al_d);
  edge_logit_max<<<gEdge, T, 0, stream>>>(ei, al_s, al_d, eval, m_enc);
  edge_exp_sum  <<<gEdge, T, 0, stream>>>(ei, eval, m_enc, denom);
  edge_scatter  <<<gScat, T, 0, stream>>>(ei, eval, denom, bufA, bufB);
  bias_elu      <<<gInit, T, 0, stream>>>(bufB, b1);                         // elu(out+b1)

  // ---- GAT layer 2: H -> H ----
  gemm64_wmma<HID, 0, 0><<<gGemm, T, 0, stream>>>(bufB, W2, nullptr, bufA, NNODES);
  init_buffers<<<gInit, T, 0, stream>>>(bufB, denom, m_enc, out, 0);         // bufB free now
  attn_logits  <<<gNode, T, 0, stream>>>(bufA, as2, ad2, al_s, al_d);
  edge_logit_max<<<gEdge, T, 0, stream>>>(ei, al_s, al_d, eval, m_enc);
  edge_exp_sum  <<<gEdge, T, 0, stream>>>(ei, eval, m_enc, denom);
  edge_scatter  <<<gScat, T, 0, stream>>>(ei, eval, denom, bufA, bufB);
  bias_elu      <<<gInit, T, 0, stream>>>(bufB, b2);

  // ---- MLP head + pooling ----
  gemm64_wmma<HID, 1, 1><<<gGemm, T, 0, stream>>>(bufB, mw1, mb1, bufA, NNODES);   // relu
  gemm64_wmma<NCLS, 0, 1><<<gGemm, T, 0, stream>>>(bufA, mw2, mb2, eval, NNODES);  // logits
  pool_kernel  <<<gNode, T, 0, stream>>>(eval, batch, out);
}
